// Attention_43963285242695
// MI455X (gfx1250) — matrix operand order repaired and checked by reading, not compiled, not run on hardware
//
#include <hip/hip_runtime.h>

// ---------------- problem constants ----------------
#define D_MODEL   4096
#define N_HEADS   32
#define KV_HEADS  8
#define HEAD_DIM  128
#define SEQ       2048
#define BATCH     2
#define TOKENS    (BATCH * SEQ)                       // 4096
#define QKV_DIM   ((KV_HEADS * 2 + N_HEADS) * HEAD_DIM)  // 6144
#define CLIP_QKV  8.0f
#define ROPE_THETA 500000.0f

typedef __attribute__((ext_vector_type(16))) _Float16 v16h;
typedef __attribute__((ext_vector_type(8)))  float    v8f;
typedef unsigned int v4u __attribute__((ext_vector_type(4)));
typedef int          v8i __attribute__((ext_vector_type(8)));
typedef int          v4i __attribute__((ext_vector_type(4)));

union Frag16 { uint4 u[2]; v16h v; };

// ---------------- WMMA fragment loaders (wave32 layouts, CDNA5 ISA) ----
// A-matrix 16x32 f16 (MxK): lane l%16 = row M; lanes 0-15 hold K 0..7 & 16..23,
// lanes 16-31 hold K 8..15 & 24..31 (2 halves packed per VGPR, ascending K).
__device__ __forceinline__ v16h load_a16(const _Float16* rowbase, int ld, int k0) {
    const int lane = threadIdx.x & 31;
    const int hi = lane >> 4, lr = lane & 15;
    Frag16 f;
    const _Float16* p = rowbase + (long)lr * ld + k0 + hi * 8;
    f.u[0] = *reinterpret_cast<const uint4*>(p);
    f.u[1] = *reinterpret_cast<const uint4*>(p + 16);
    return f.v;
}

// B-matrix 32x16 f16 (KxN): lane l%16 = column N; lanes 0-15 hold K 0..7 & 16..23,
// lanes 16-31 hold K 8..15 & 24..31 (2 K per VGPR). Source stored [N][K] row-major.
__device__ __forceinline__ v16h load_b16(const _Float16* rowbase, long ld, int k0) {
    const int lane = threadIdx.x & 31;
    const int hi = lane >> 4, lr = lane & 15;
    Frag16 f;
    const _Float16* p = rowbase + (long)lr * ld + k0 + hi * 8;
    f.u[0] = *reinterpret_cast<const uint4*>(p);
    f.u[1] = *reinterpret_cast<const uint4*>(p + 16);
    return f.v;
}

__device__ __forceinline__ v8f wmma_f16(v16h a, v16h b, v8f c) {
    return __builtin_amdgcn_wmma_f32_16x16x32_f16(false, a, false, b, (short)0, c,
                                                  false, false);
}

// ---------------- Tensor Data Mover: 2D tile (f16) global -> padded LDS -----------
// D# per CDNA5 ISA. data_size=2B. LDS padding:
// pad_interval code 4 = every 32 DWORDs (64 halves), pad_amount code 3 = 4 DWORDs
// (8 halves) -> LDS row stride 72 halves, matching the WMMA fragment readers.
// clang-23 toolchain: 6-arg builtin (uint32x4, int32x8, int32x4, int32x4, int32x8, i32).
__device__ __forceinline__ void tdm_load_2d(const _Float16* gsrc, unsigned lds_off,
                                            unsigned tensor_d0, unsigned tensor_d1,
                                            unsigned long long stride0_elems,
                                            unsigned tile_d0, unsigned tile_d1) {
    const unsigned long long ga = (unsigned long long)(const void*)gsrc;
    v4u g0;
    g0[0] = 1u;                                            // count=1, user descriptor
    g0[1] = lds_off;                                       // LDS byte address
    g0[2] = (unsigned)(ga & 0xffffffffu);                  // global_addr[31:0]
    g0[3] = (unsigned)((ga >> 32) & 0x01ffffffu) | (2u << 30);  // ga[56:32], type=2
    v8i g1;
    g1[0] = (int)((1u << 16) |            // data_size = 1 -> 2 bytes
                  (1u << 20) |            // pad_enable
                  (4u << 22) |            // pad_interval: 32 DWORDs
                  (3u << 25));            // pad_amount: 4 DWORDs
    g1[1] = (int)((tensor_d0 & 0xffffu) << 16);                    // td0[15:0]
    g1[2] = (int)(((tensor_d0 >> 16) & 0xffffu) |
                  ((tensor_d1 & 0xffffu) << 16));                  // td0[31:16], td1[15:0]
    g1[3] = (int)(((tensor_d1 >> 16) & 0xffffu) |
                  ((tile_d0 & 0xffffu) << 16));                    // td1[31:16], tile_d0
    g1[4] = (int)(tile_d1 & 0xffffu);                              // tile_d1, tile_d2=0
    g1[5] = (int)(unsigned)(stride0_elems & 0xffffffffu);          // stride0[31:0]
    g1[6] = (int)((stride0_elems >> 32) & 0xffffu);                // stride0[47:32]
    g1[7] = 0;
    const v4i g2 = {0, 0, 0, 0};             // 2-D tensor: groups 2/3 unused
    const v4i g3 = {0, 0, 0, 0};
    const v8i g4 = {0, 0, 0, 0, 0, 0, 0, 0}; // extra group (clang-23 form), unused
    __builtin_amdgcn_tensor_load_to_lds(g0, g1, g2, g3, g4, 0);
}

// ---------------- fp32 -> f16 staging (halves HBM traffic for all GEMM operands) --
__global__ __launch_bounds__(256) void cvt_f32_to_f16(const float* __restrict__ src,
                                                      _Float16* __restrict__ dst,
                                                      long n) {
    long i = ((long)blockIdx.x * 256 + threadIdx.x) * 4;
    if (i + 3 < n) {
        float4 f = *reinterpret_cast<const float4*>(src + i);
        union { _Float16 h[4]; uint2 u; } o;
        o.h[0] = (_Float16)f.x; o.h[1] = (_Float16)f.y;
        o.h[2] = (_Float16)f.z; o.h[3] = (_Float16)f.w;
        *reinterpret_cast<uint2*>(dst + i) = o.u;
    }
}

// ---------------- LDS-tiled WMMA GEMM: C[M,N] = A[M,K] * B[N,K]^T, K = 4096 -------
// Block = 256 threads (8 waves), tile 64(M) x 128(N) x 64(K).
// Tiles staged by the Tensor Data Mover (double buffered; wave 0 issues, per-wave
// TENSORcnt waited by wave 0 then published via barrier). Wave (wm,wn) computes
// rows wm*16..+16, cols wn*64..+64.
// EPI 0: clip to +-8, store f16 (QKV). EPI 1: store fp32 (final output).
template <int NOUT, int EPI>
__global__ __launch_bounds__(256) void gemm_wmma(const _Float16* __restrict__ A,
                                                 const _Float16* __restrict__ Bw,
                                                 _Float16* __restrict__ outH,
                                                 float* __restrict__ outF) {
    __shared__ __align__(16) _Float16 As[2][64][72];   // 72 = 64 + TDM pad (8 halves)
    __shared__ __align__(16) _Float16 Bs[2][128][72];
    const int tid  = threadIdx.x;
    const int lane = tid & 31;
    const int wv   = tid >> 5;
    const int wm   = wv & 3;
    const int wn   = wv >> 2;
    const int hi   = lane >> 4, lr = lane & 15;
    const long m_blk = (long)blockIdx.y * 64;
    const long n_blk = (long)blockIdx.x * 128;

    const unsigned ldsA[2] = {
        (unsigned)(unsigned long long)(const void*)&As[0][0][0],
        (unsigned)(unsigned long long)(const void*)&As[1][0][0]};
    const unsigned ldsB[2] = {
        (unsigned)(unsigned long long)(const void*)&Bs[0][0][0],
        (unsigned)(unsigned long long)(const void*)&Bs[1][0][0]};

    const v8f zf = {0.f, 0.f, 0.f, 0.f, 0.f, 0.f, 0.f, 0.f};
    v8f acc[4] = {zf, zf, zf, zf};

    const int NT = 4096 / 64;
    if (wv == 0) {  // prologue: DMA tile 0 into buffer 0
        tdm_load_2d(A  + m_blk * 4096, ldsA[0], 4096, TOKENS, 4096, 64, 64);
        tdm_load_2d(Bw + n_blk * 4096, ldsB[0], 4096, NOUT,   4096, 64, 128);
    }

    for (int kt = 0; kt < NT; ++kt) {
        const int buf = kt & 1;
        if (wv == 0) __builtin_amdgcn_s_wait_tensorcnt(0);  // buf ready (wave 0)
        __syncthreads();                                     // publish LDS to all waves
        if (wv == 0 && kt + 1 < NT) {                        // DMA next tile -> buf^1
            const int k0n = (kt + 1) * 64;
            tdm_load_2d(A  + m_blk * 4096 + k0n, ldsA[buf ^ 1], 4096, TOKENS, 4096,
                        64, 64);
            tdm_load_2d(Bw + n_blk * 4096 + k0n, ldsB[buf ^ 1], 4096, NOUT, 4096,
                        64, 128);
        }
#pragma unroll
        for (int kk = 0; kk < 2; ++kk) {
            v16h a = load_a16(&As[buf][wm * 16][0], 72, kk * 32);
#pragma unroll
            for (int f = 0; f < 4; ++f) {
                v16h b = load_b16(&Bs[buf][wn * 64 + f * 16][0], 72, kk * 32);
                acc[f] = wmma_f16(a, b, acc[f]);
            }
        }
    }
    // epilogue: C layout -> VGPR r holds rows r (lanes 0-15) / r+8 (lanes 16-31)
#pragma unroll
    for (int f = 0; f < 4; ++f) {
        const long col = n_blk + wn * 64 + f * 16 + lr;
#pragma unroll
        for (int r = 0; r < 8; ++r) {
            const long row = m_blk + wm * 16 + r + hi * 8;
            float v = acc[f][r];
            if (EPI == 0) {
                v = fminf(CLIP_QKV, fmaxf(-CLIP_QKV, v));
                outH[row * NOUT + col] = (_Float16)v;
            } else {
                outF[row * NOUT + col] = v;
            }
        }
    }
}

// ---------------- RoPE + scatter into Q [b,h,L,d], K [b,kh,L,d], V^T [b,kh,d,L] ---
__global__ __launch_bounds__(256) void rope_scatter(const _Float16* __restrict__ qkv,
                                                    _Float16* __restrict__ Qb,
                                                    _Float16* __restrict__ Kb,
                                                    _Float16* __restrict__ Vt) {
    const long QN = (long)TOKENS * N_HEADS * 64;   // rope pairs for Q
    const long KN = (long)TOKENS * KV_HEADS * 64;  // rope pairs for K
    const long VN = (long)TOKENS * KV_HEADS * 128; // elements of V
    long id = (long)blockIdx.x * 256 + threadIdx.x;
    if (id < QN) {
        int  p   = id & 63;
        long r   = id >> 6;
        int  h   = r & 31;
        long t   = r >> 5;
        int  b   = (int)(t >> 11);
        int  pos = (int)(t & 2047);
        float q1 = (float)qkv[t * QKV_DIM + (long)h * HEAD_DIM + p];
        float q2 = (float)qkv[t * QKV_DIM + (long)h * HEAD_DIM + p + 64];
        float inv_freq = __powf(ROPE_THETA, -(float)p * (1.0f / 64.0f));
        float sn, cs;
        __sincosf((float)pos * inv_freq, &sn, &cs);
        long ob = ((long)(b * N_HEADS + h) * SEQ + pos) * HEAD_DIM;
        Qb[ob + p]      = (_Float16)(q1 * cs - q2 * sn);
        Qb[ob + p + 64] = (_Float16)(q2 * cs + q1 * sn);
    } else if (id < QN + KN) {
        long i2  = id - QN;
        int  p   = i2 & 63;
        long r   = i2 >> 6;
        int  kh  = r & 7;
        long t   = r >> 3;
        int  b   = (int)(t >> 11);
        int  pos = (int)(t & 2047);
        float k1 = (float)qkv[t * QKV_DIM + D_MODEL + (long)kh * HEAD_DIM + p];
        float k2 = (float)qkv[t * QKV_DIM + D_MODEL + (long)kh * HEAD_DIM + p + 64];
        float inv_freq = __powf(ROPE_THETA, -(float)p * (1.0f / 64.0f));
        float sn, cs;
        __sincosf((float)pos * inv_freq, &sn, &cs);
        long ob = ((long)(b * KV_HEADS + kh) * SEQ + pos) * HEAD_DIM;
        Kb[ob + p]      = (_Float16)(k1 * cs - k2 * sn);
        Kb[ob + p + 64] = (_Float16)(k2 * cs + k1 * sn);
    } else if (id < QN + KN + VN) {
        long i2  = id - QN - KN;
        int  d   = i2 & 127;
        long r   = i2 >> 7;
        int  kh  = r & 7;
        long t   = r >> 3;
        int  b   = (int)(t >> 11);
        int  pos = (int)(t & 2047);
        // V transposed: [b][kh][d][pos] so PV WMMA B-frags are contiguous in key dim
        Vt[((long)(b * KV_HEADS + kh) * HEAD_DIM + d) * SEQ + pos] =
            qkv[t * QKV_DIM + D_MODEL + KV_HEADS * HEAD_DIM + (long)kh * HEAD_DIM + d];
    }
}

// ---------------- flash attention: one wave per (b, h, 16-query tile) -------------
// Online softmax; per 32-key chunk: 8 QK^T WMMAs + 8 PV WMMAs. P re-laid out
// C-layout -> A-layout via a 16x40 padded LDS tile (single wave: LDS in-order).
__global__ __launch_bounds__(32) void flash_attn(const _Float16* __restrict__ Q,
                                                 const _Float16* __restrict__ Kst,
                                                 const _Float16* __restrict__ Vt,
                                                 _Float16* __restrict__ attn) {
    __shared__ __align__(16) _Float16 Ps[16][40];
    const int lane = threadIdx.x & 31;
    const int hi = lane >> 4, lr = lane & 15;
    const int qt = blockIdx.x & 127;
    const int h  = (blockIdx.x >> 7) & 31;
    const int b  = blockIdx.x >> 12;
    const int kh = h >> 2;   // GQA: 4 query heads per kv head
    const int q0 = qt * 16;

    const _Float16* Qbase = Q   + ((long)(b * N_HEADS  + h ) * SEQ + q0) * HEAD_DIM;
    const _Float16* Kbase = Kst + ((long)(b * KV_HEADS + kh) * SEQ) * HEAD_DIM;
    const _Float16* Vbase = Vt  + ((long)(b * KV_HEADS + kh) * HEAD_DIM) * SEQ;

    v16h qf[4];
#pragma unroll
    for (int c = 0; c < 4; ++c) qf[c] = load_a16(Qbase, HEAD_DIM, c * 32);

    const v8f zf = {0.f, 0.f, 0.f, 0.f, 0.f, 0.f, 0.f, 0.f};
    v8f o[8];
    float m[8], l[8];
#pragma unroll
    for (int n = 0; n < 8; ++n) o[n] = zf;
#pragma unroll
    for (int r = 0; r < 8; ++r) { m[r] = -1e30f; l[r] = 0.f; }

    const int   nch = (q0 + 47) >> 5;             // causal: chunks of 32 keys
    const float sc  = 0.08838834764831845f;       // 1/sqrt(128)

    for (int ck = 0; ck < nch; ++ck) {
        const int kb = ck * 32;
        v8f s0 = zf, s1 = zf;
#pragma unroll
        for (int c = 0; c < 4; ++c) {             // contract over head_dim
            v16h kb0 = load_b16(Kbase + (long)kb * HEAD_DIM,        HEAD_DIM, c * 32);
            v16h kb1 = load_b16(Kbase + (long)(kb + 16) * HEAD_DIM, HEAD_DIM, c * 32);
            s0 = wmma_f16(qf[c], kb0, s0);
            s1 = wmma_f16(qf[c], kb1, s1);
        }
        // online softmax update (rows live in lane halves; reduce across 16 lanes)
#pragma unroll
        for (int r = 0; r < 8; ++r) {
            const int qrow = q0 + r + hi * 8;
            float v0 = s0[r] * sc;
            float v1 = s1[r] * sc;
            if (kb + lr > qrow)      v0 = -1e30f;
            if (kb + 16 + lr > qrow) v1 = -1e30f;
            float mx = fmaxf(v0, v1);
#pragma unroll
            for (int offx = 8; offx >= 1; offx >>= 1)
                mx = fmaxf(mx, __shfl_xor(mx, offx, 32));
            const float mn = fmaxf(m[r], mx);
            const float al = __expf(m[r] - mn);
            const float p0 = __expf(v0 - mn);
            const float p1 = __expf(v1 - mn);
            float rs = p0 + p1;
#pragma unroll
            for (int offx = 8; offx >= 1; offx >>= 1)
                rs += __shfl_xor(rs, offx, 32);
            l[r] = l[r] * al + rs;
            m[r] = mn;
#pragma unroll
            for (int n = 0; n < 8; ++n) o[n][r] *= al;
            Ps[r + hi * 8][lr]      = (_Float16)p0;
            Ps[r + hi * 8][16 + lr] = (_Float16)p1;
        }
        __syncthreads();
        v16h pf = load_a16(&Ps[0][0], 40, 0);
#pragma unroll
        for (int n = 0; n < 8; ++n) {             // P (16x32) x V^T chunk (32x16)
            v16h vf = load_b16(Vbase + (long)(n * 16) * SEQ, SEQ, kb);
            o[n] = wmma_f16(pf, vf, o[n]);
        }
        __syncthreads();
    }
    // normalize + write [tok, h*128 + d] (f16 for the final GEMM)
#pragma unroll
    for (int r = 0; r < 8; ++r) {
        const float inv = 1.0f / l[r];
        const long  tok = (long)b * SEQ + q0 + r + hi * 8;
#pragma unroll
        for (int n = 0; n < 8; ++n) {
            attn[tok * D_MODEL + (long)h * HEAD_DIM + n * 16 + lr] =
                (_Float16)(o[n][r] * inv);
        }
    }
}

// ---------------- launch -----------------------------------------------------------
extern "C" void kernel_launch(void* const* d_in, const int* in_sizes, int n_in,
                              void* d_out, int out_size, void* d_ws, size_t ws_size,
                              hipStream_t stream) {
    const float* x    = (const float*)d_in[0];
    const float* Wqkv = (const float*)d_in[1];
    const float* Wout = (const float*)d_in[2];
    float* out = (float*)d_out;

    char* ws = (char*)d_ws;
    size_t off = 0;
    auto alloc16 = [&](size_t elems) {
        _Float16* p = (_Float16*)(ws + off);
        off += elems * sizeof(_Float16);
        return p;
    };
    _Float16* x16    = alloc16((size_t)TOKENS * D_MODEL);
    _Float16* wq16   = alloc16((size_t)QKV_DIM * D_MODEL);
    _Float16* wo16   = alloc16((size_t)D_MODEL * D_MODEL);
    _Float16* qkv16  = alloc16((size_t)TOKENS * QKV_DIM);
    _Float16* Qb     = alloc16((size_t)BATCH * N_HEADS * SEQ * HEAD_DIM);
    _Float16* Kb     = alloc16((size_t)BATCH * KV_HEADS * SEQ * HEAD_DIM);
    _Float16* Vt     = alloc16((size_t)BATCH * KV_HEADS * HEAD_DIM * SEQ);
    _Float16* attn16 = alloc16((size_t)TOKENS * D_MODEL);

    const long nx = (long)TOKENS * D_MODEL;
    const long nq = (long)QKV_DIM * D_MODEL;
    const long no = (long)D_MODEL * D_MODEL;
    cvt_f32_to_f16<<<(nx / 4 + 255) / 256, 256, 0, stream>>>(x, x16, nx);
    cvt_f32_to_f16<<<(nq / 4 + 255) / 256, 256, 0, stream>>>(Wqkv, wq16, nq);
    cvt_f32_to_f16<<<(no / 4 + 255) / 256, 256, 0, stream>>>(Wout, wo16, no);

    // QKV projection + clip
    gemm_wmma<QKV_DIM, 0><<<dim3(QKV_DIM / 128, TOKENS / 64), 256, 0, stream>>>(
        x16, wq16, qkv16, nullptr);

    // RoPE + scatter into attention layouts
    const long total = (long)TOKENS * N_HEADS * 64 + (long)TOKENS * KV_HEADS * 64 +
                       (long)TOKENS * KV_HEADS * 128;
    rope_scatter<<<(unsigned)((total + 255) / 256), 256, 0, stream>>>(qkv16, Qb, Kb, Vt);

    // causal GQA flash attention
    flash_attn<<<BATCH * N_HEADS * (SEQ / 16), 32, 0, stream>>>(Qb, Kb, Vt, attn16);

    // output projection -> fp32
    gemm_wmma<D_MODEL, 1><<<dim3(D_MODEL / 128, TOKENS / 64), 256, 0, stream>>>(
        attn16, wo16, nullptr, out);
}
